// StrassenAttentionWithALiBi_61993557950733
// MI455X (gfx1250) — matrix operand order repaired and checked by reading, not compiled, not run on hardware
//
#include <hip/hip_runtime.h>
#include <hip/hip_bf16.h>

// ---------------- problem constants ----------------
#define Bb  2
#define Nn  256
#define Cc  512
#define Hh  8
#define Dd  64
#define BH  16          // Bb*Hh
#define M5  2560        // 5*Cc
#define MR  512         // Bb*Nn rows of projection
#define SCALER 0.125f   // 1/sqrt(64)
#define YVS 264         // padded LDS row stride for Yv (256 + 8 elems = 528 B)

typedef __attribute__((ext_vector_type(16))) __bf16 v16bf;
typedef __attribute__((ext_vector_type(8)))  __bf16 v8bf;
typedef __attribute__((ext_vector_type(8)))  float  v8f;
typedef __attribute__((ext_vector_type(4)))  unsigned int u32x4;
typedef __attribute__((ext_vector_type(8)))  int i32x8;
typedef __attribute__((ext_vector_type(4)))  int i32x4;

static __device__ __forceinline__ v8f wmma_bf16(v16bf a, v16bf b, v8f c) {
    return __builtin_amdgcn_wmma_f32_16x16x32_bf16(
        false, a, false, b, (short)0, c, false, false);
}

// A-operand (16x32 bf16 MxK): lane&15=row; e -> K = 16*(lane>>4) + e
// -> two contiguous 16-byte runs: [kb, kb+8) and [kb+8, kb+16)
static __device__ __forceinline__ v16bf loadA16v(const __bf16* rowp, int lane) {
    int kb = (lane >> 4) << 4;
    v8bf lo = *(const v8bf*)(rowp + kb);
    v8bf hi = *(const v8bf*)(rowp + kb + 8);
    v16bf a;
#pragma unroll
    for (int e = 0; e < 8; ++e) { a[e] = lo[e]; a[e + 8] = hi[e]; }
    return a;
}
// B-operand (32x16 bf16 KxN): lane&15=col; e -> K = 16*(lane>>4)+e  (contiguous 32 B)
static __device__ __forceinline__ v16bf loadB16v(const __bf16* colp, int lane) {
    return *(const v16bf*)(colp + ((lane >> 4) << 4));
}

// ---- Tensor Data Mover: load one strided column (256 rows x 2 B, stride 128 B)
// into LDS.  D# per ISA 8.3/8.4: count=1, type=2, data_size=2B,
// tensor_dim0=64, tensor_dim1=256, tile_dim0=1, tile_dim1=256, dim0_stride=64.
static __device__ __forceinline__ void tdm_load_col(const __bf16* gcol,
                                                    unsigned lds_off) {
    unsigned long long ga = (unsigned long long)(uintptr_t)gcol;
    u32x4 g0 = { 1u, lds_off, (unsigned)ga,
                 (unsigned)((ga >> 32) & 0x1FFFFFFu) | (2u << 30) };
    i32x8 g1 = { 0x00010000,      // data_size=1 (2 bytes), wg_mask=0
                 (64 << 16),      // tensor_dim0[15:0]=64 at bits[63:48]
                 (256 << 16),     // tensor_dim1[15:0]=256 at bits[95:80]
                 (1 << 16),       // tile_dim0=1 at bits[127:112]
                 256,             // tile_dim1=256 at bits[143:128]
                 64,              // tensor_dim0_stride[31:0]=64
                 0, 0 };
    i32x4 z4 = { 0, 0, 0, 0 };
#if defined(__clang_major__) && (__clang_major__ >= 23)
    i32x8 z8 = { 0, 0, 0, 0, 0, 0, 0, 0 };
    __builtin_amdgcn_tensor_load_to_lds(g0, g1, z4, z4, z8, 0);
#else
    __builtin_amdgcn_tensor_load_to_lds(g0, g1, z4, z4, 0);
#endif
}

// ============ kernel 1: w = hidden @ W.T -> qkv[5][BH][N][D] bf16 ============
// grid (MR/16, M5/16), block 32
__global__ void k_proj(const float* __restrict__ hs, const float* __restrict__ W,
                       __bf16* __restrict__ qkv) {
    int lane = threadIdx.x, half = lane >> 4, l15 = lane & 15;
    int mt = blockIdx.x, ot = blockIdx.y;
    v8f acc = {};
    const float* arow = hs + (size_t)(mt * 16 + l15) * Cc;
    const float* brow = W  + (size_t)(ot * 16 + l15) * Cc;
    for (int kt = 0; kt < Cc; kt += 32) {
        int ka = kt + (half << 4), kb = kt + (half << 4);
        v8f alo = *(const v8f*)(arow + ka);
        v8f ahi = *(const v8f*)(arow + ka + 8);
        v8f blo = *(const v8f*)(brow + kb);
        v8f bhi = *(const v8f*)(brow + kb + 8);
        v16bf a, b;
#pragma unroll
        for (int e = 0; e < 8; ++e) {
            a[e] = (__bf16)alo[e]; a[e + 8] = (__bf16)ahi[e];
            b[e] = (__bf16)blo[e]; b[e + 8] = (__bf16)bhi[e];
        }
        acc = wmma_bf16(a, b, acc);
    }
    int o = ot * 16 + l15;          // output column (row of W)
    int s = o >> 9, rem = o & 511;
    int h = rem >> 6, d = rem & 63;
    size_t sbase = (size_t)s * (BH * Nn * Dd);
#pragma unroll
    for (int r = 0; r < 8; ++r) {
        int m = mt * 16 + (half << 3) + r;
        int b_ = m >> 8, n_ = m & 255;
        qkv[sbase + (((size_t)(b_ * Hh + h)) * Nn + n_) * Dd + d] = (__bf16)acc[r];
    }
}

// ============ kernel 2: Xs[i][j], Ys[j][k], Zt[i][k] (f32 + pos bias) ============
// grid (BH, N/16, N/16), block 32
__global__ void k_scores(const __bf16* __restrict__ abf, const __bf16* __restrict__ bbf,
                         const __bf16* __restrict__ cbf,
                         float* Xs, float* Ys, float* Zts) {
    int lane = threadIdx.x, half = lane >> 4, l15 = lane & 15;
    int bh = blockIdx.x, rt = blockIdx.y, ct = blockIdx.z;
    size_t qb = (size_t)bh * Nn * Dd;
    const __bf16* aR = abf + qb + (size_t)(rt * 16 + l15) * Dd;
    const __bf16* bR = bbf + qb + (size_t)(rt * 16 + l15) * Dd;
    const __bf16* bC = bbf + qb + (size_t)(ct * 16 + l15) * Dd;
    const __bf16* cC = cbf + qb + (size_t)(ct * 16 + l15) * Dd;
    v8f x = {}, y = {}, z = {};
#pragma unroll
    for (int kt = 0; kt < Dd; kt += 32) {
        v16bf aA = loadA16v(aR + kt, lane);
        v16bf bA = loadA16v(bR + kt, lane);
        v16bf bB = loadB16v(bC + kt, lane);
        v16bf cB = loadB16v(cC + kt, lane);
        x = wmma_bf16(aA, bB, x);   // a . bq
        y = wmma_bf16(bA, cB, y);   // bq . cq
        z = wmma_bf16(aA, cB, z);   // a . cq  (= Z transposed)
    }
    size_t tb = (size_t)bh * Nn * Nn;
#pragma unroll
    for (int r = 0; r < 8; ++r) {
        int row = rt * 16 + (half << 3) + r;
        int col = ct * 16 + l15;
        float pbf = fminf((float)(col - row), 0.0f);
        float pbr = fminf((float)(row - col), 0.0f);
        size_t idx = tb + (size_t)row * Nn + col;
        Xs[idx]  = x[r] * SCALER + pbf;
        Ys[idx]  = y[r] * SCALER + pbf;
        Zts[idx] = z[r] * SCALER + pbr;
    }
}

// ============ kernel 3a: per-row max ============  grid (BH, N), block 256
__global__ void k_rowmax(const float* __restrict__ src, float* dst) {
    int bh = blockIdx.x, row = blockIdx.y;
    float v = src[((size_t)bh * Nn + row) * Nn + threadIdx.x];
#pragma unroll
    for (int off = 16; off; off >>= 1) v = fmaxf(v, __shfl_xor(v, off, 32));
    __shared__ float s[8];
    if ((threadIdx.x & 31) == 0) s[threadIdx.x >> 5] = v;
    __syncthreads();
    if (threadIdx.x == 0) {
        float m = s[0];
#pragma unroll
        for (int i = 1; i < 8; ++i) m = fmaxf(m, s[i]);
        dst[bh * Nn + row] = m;
    }
}

// ============ kernel 3b: per-head max ============  grid BH, block 256
__global__ void k_bhmax(const float* __restrict__ src, float* dst) {
    int bh = blockIdx.x;
    const float* p = src + (size_t)bh * Nn * Nn;
    float v = -3.4e38f;
    for (int i = threadIdx.x; i < Nn * Nn; i += 256) v = fmaxf(v, p[i]);
#pragma unroll
    for (int off = 16; off; off >>= 1) v = fmaxf(v, __shfl_xor(v, off, 32));
    __shared__ float s[8];
    if ((threadIdx.x & 31) == 0) s[threadIdx.x >> 5] = v;
    __syncthreads();
    if (threadIdx.x == 0) {
        float m = s[0];
#pragma unroll
        for (int i = 1; i < 8; ++i) m = fmaxf(m, s[i]);
        dst[bh] = m;
    }
}

// ============ kernel 4: exp + bf16 convert (+YeT) ============ grid (BH,N), block 256
__global__ void k_expcvt(const float* Xs, const float* Ys, const float* Zts,
                         const float* Xmax, const float* Ymax, const float* Zmax,
                         __bf16* Xe, __bf16* Ye, __bf16* YeT, __bf16* Zte) {
    int bh = blockIdx.x, row = blockIdx.y, t = threadIdx.x;
    size_t idx = ((size_t)bh * Nn + row) * Nn + t;
    Xe[idx]  = (__bf16)__expf(Xs[idx] - Xmax[bh * Nn + row]);
    float ye = __expf(Ys[idx] - Ymax[bh]);
    Ye[idx]  = (__bf16)ye;
    YeT[((size_t)bh * Nn + t) * Nn + row] = (__bf16)ye;
    Zte[idx] = (__bf16)__expf(Zts[idx] - Zmax[bh * Nn + row]);
}

// ============ kernel 5: down[i] = sum_j X[i,j] * (sum_k Zt[i,k] Y[j,k]) ============
// grid (BH, N/16), block 32
__global__ void k_down(const __bf16* __restrict__ Zte, const __bf16* __restrict__ Ye,
                       const __bf16* __restrict__ Xe, float* down) {
    int lane = threadIdx.x, half = lane >> 4, l15 = lane & 15;
    int bh = blockIdx.x, it = blockIdx.y;
    size_t tb = (size_t)bh * Nn * Nn;
    float dn[8] = {0, 0, 0, 0, 0, 0, 0, 0};
    const __bf16* aRow = Zte + tb + (size_t)(it * 16 + l15) * Nn;
    for (int jt = 0; jt < 16; ++jt) {
        const __bf16* bRow = Ye + tb + (size_t)(jt * 16 + l15) * Nn;
        v8f u = {};
#pragma unroll
        for (int ks = 0; ks < Nn; ks += 32)
            u = wmma_bf16(loadA16v(aRow + ks, lane), loadB16v(bRow + ks, lane), u);
#pragma unroll
        for (int r = 0; r < 8; ++r) {
            int row = it * 16 + (half << 3) + r;
            int col = jt * 16 + l15;
            dn[r] += u[r] * (float)Xe[tb + (size_t)row * Nn + col];
        }
    }
#pragma unroll
    for (int r = 0; r < 8; ++r) {
        float v = dn[r];
#pragma unroll
        for (int off = 1; off < 16; off <<= 1) v += __shfl_xor(v, off, 16);
        if (l15 == 0) down[bh * Nn + it * 16 + (half << 3) + r] = v;
    }
}

// ============ kernel 6: main contraction, one (bh,d) per block ============
// up[i,d] = sum_k (Xe @ (Ye*v1_d))[i,k] * Zt[i,k] * v2[k,d]
// grid (BH, D), block 256 (8 waves); dynamic LDS: Yv[256][YVS] + v1s + v2s (bf16)
__global__ void k_main(const __bf16* __restrict__ Xe, const __bf16* __restrict__ YeT,
                       const __bf16* __restrict__ Zte,
                       const __bf16* __restrict__ v1bf, const __bf16* __restrict__ v2bf,
                       float* up) {
    extern __shared__ __bf16 smem[];
    __bf16* Yv  = smem;                 // [Nn][YVS]
    __bf16* v1s = smem + Nn * YVS;      // [Nn]
    __bf16* v2s = v1s + Nn;             // [Nn]

    int bh = blockIdx.x, d = blockIdx.y, t = threadIdx.x;
    size_t tb = (size_t)bh * Nn * Nn;
    size_t vb = (size_t)bh * Nn * Dd + d;

    // --- stage v1[:,d] and v2[:,d] via the Tensor Data Mover (wave 0 only) ---
    if (t < 32) {
        tdm_load_col(v1bf + vb, (unsigned)(uintptr_t)v1s);
        tdm_load_col(v2bf + vb, (unsigned)(uintptr_t)v2s);
        __builtin_amdgcn_s_wait_tensorcnt(0);
    }
    __syncthreads();

    // --- build Yv[k][j] = YeT[k][j] * v1[j,d] in LDS, one row per thread ---
    {
        const __bf16* yrow = YeT + tb + (size_t)t * Nn;
        __bf16* dst = Yv + (size_t)t * YVS;
#pragma unroll
        for (int j0 = 0; j0 < Nn; j0 += 16) {
            v16bf yv = *(const v16bf*)(yrow + j0);
            v16bf o;
#pragma unroll
            for (int e = 0; e < 16; ++e)
                o[e] = (__bf16)((float)yv[e] * (float)v1s[j0 + e]);
            *(v16bf*)(dst + j0) = o;
        }
    }
    __syncthreads();

    int wave = t >> 5, lane = t & 31, half = lane >> 4, l15 = lane & 15;
    const __bf16* a0base = Xe + tb + (size_t)(wave * 16 + l15) * Nn;
    const __bf16* a1base = Xe + tb + (size_t)((wave + 8) * 16 + l15) * Nn;

    float up0[8] = {0, 0, 0, 0, 0, 0, 0, 0};
    float up1[8] = {0, 0, 0, 0, 0, 0, 0, 0};

    for (int kt = 0; kt < 16; ++kt) {
        int kc = kt * 16 + l15;                       // this lane's k column
        const __bf16* bl = Yv + (size_t)kc * YVS;     // LDS row (Yv[kc][:])
        v8f t0 = {}, t1 = {};
#pragma unroll 2
        for (int js = 0; js < Nn; js += 32) {
            __builtin_prefetch(a0base + js + 64, 0, 0);   // global_prefetch_b8
            v16bf bvec = *(const v16bf*)(bl + js + ((lane >> 4) << 4)); // ds_load_b128 x2
            v16bf a0 = loadA16v(a0base + js, lane);
            v16bf a1 = loadA16v(a1base + js, lane);
            t0 = wmma_bf16(a0, bvec, t0);
            t1 = wmma_bf16(a1, bvec, t1);
        }
        float v2k = (float)v2s[kc];
#pragma unroll
        for (int r = 0; r < 8; ++r) {
            int m = (half << 3) + r;
            int i0 = wave * 16 + m, i1 = (wave + 8) * 16 + m;
            up0[r] += t0[r] * (float)Zte[tb + (size_t)i0 * Nn + kc] * v2k;
            up1[r] += t1[r] * (float)Zte[tb + (size_t)i1 * Nn + kc] * v2k;
        }
    }
#pragma unroll
    for (int r = 0; r < 8; ++r) {
        float v = up0[r];
#pragma unroll
        for (int off = 1; off < 16; off <<= 1) v += __shfl_xor(v, off, 16);
        if (l15 == 0)
            up[((size_t)bh * Nn + wave * 16 + (half << 3) + r) * Dd + d] = v;
        float w2 = up1[r];
#pragma unroll
        for (int off = 1; off < 16; off <<= 1) w2 += __shfl_xor(w2, off, 16);
        if (l15 == 0)
            up[((size_t)bh * Nn + (wave + 8) * 16 + (half << 3) + r) * Dd + d] = w2;
    }
}

// ============ kernel 7: out = up / (down + 1e-9), head transpose ============
__global__ void k_out(const float* __restrict__ up, const float* __restrict__ down,
                      float* out) {
    int idx = blockIdx.x * 256 + threadIdx.x;
    int d = idx & 63, h = (idx >> 6) & 7, n = (idx >> 9) & 255, b_ = idx >> 17;
    int bh = b_ * 8 + h;
    out[idx] = up[((size_t)bh * Nn + n) * Dd + d] / (down[bh * Nn + n] + 1e-9f);
}

// =============================== launch ===============================
extern "C" void kernel_launch(void* const* d_in, const int* in_sizes, int n_in,
                              void* d_out, int out_size, void* d_ws, size_t ws_size,
                              hipStream_t stream) {
    const float* hs = (const float*)d_in[0];   // [B,N,C] f32
    const float* W  = (const float*)d_in[1];   // [5C,C] f32
    float* out = (float*)d_out;                // [B,N,C] f32

    char* ws = (char*)d_ws;
    size_t off = 0;
    auto alloc = [&](size_t bytes) -> void* {
        void* p = ws + off;
        off += (bytes + 255) & ~(size_t)255;
        return p;
    };
    const size_t qkvB = (size_t)BH * Nn * Dd * sizeof(__bf16);   // 512 KB
    const size_t nnF  = (size_t)BH * Nn * Nn * sizeof(float);    // 4 MB
    const size_t nnB  = (size_t)BH * Nn * Nn * sizeof(__bf16);   // 2 MB

    __bf16* qkv  = (__bf16*)alloc(5 * qkvB);   // a | bq | cq | v1 | v2 contiguous
    __bf16* abf  = qkv;
    __bf16* bbf  = qkv + 1 * (BH * Nn * Dd);
    __bf16* cbf  = qkv + 2 * (BH * Nn * Dd);
    __bf16* v1bf = qkv + 3 * (BH * Nn * Dd);
    __bf16* v2bf = qkv + 4 * (BH * Nn * Dd);
    float*  Xs   = (float*)alloc(nnF);
    float*  Ys   = (float*)alloc(nnF);
    float*  Zts  = (float*)alloc(nnF);
    __bf16* Xe   = (__bf16*)alloc(nnB);
    __bf16* Ye   = (__bf16*)alloc(nnB);
    __bf16* YeT  = (__bf16*)alloc(nnB);
    __bf16* Zte  = (__bf16*)alloc(nnB);
    float*  Xmax = (float*)alloc((size_t)BH * Nn * sizeof(float));
    float*  Ymax = (float*)alloc((size_t)BH * sizeof(float));
    float*  Zmax = (float*)alloc((size_t)BH * Nn * sizeof(float));
    float*  upB  = (float*)alloc((size_t)BH * Nn * Dd * sizeof(float));
    float*  dnB  = (float*)alloc((size_t)BH * Nn * sizeof(float));
    (void)ws_size; (void)in_sizes; (void)n_in; (void)out_size;

    const unsigned mainSmem = (unsigned)((Nn * YVS + 2 * Nn) * sizeof(__bf16));

    k_proj  <<<dim3(MR / 16, M5 / 16), 32, 0, stream>>>(hs, W, qkv);
    k_scores<<<dim3(BH, Nn / 16, Nn / 16), 32, 0, stream>>>(abf, bbf, cbf, Xs, Ys, Zts);
    k_rowmax<<<dim3(BH, Nn), 256, 0, stream>>>(Xs, Xmax);
    k_rowmax<<<dim3(BH, Nn), 256, 0, stream>>>(Zts, Zmax);
    k_bhmax <<<BH, 256, 0, stream>>>(Ys, Ymax);
    k_expcvt<<<dim3(BH, Nn), 256, 0, stream>>>(Xs, Ys, Zts, Xmax, Ymax, Zmax,
                                               Xe, Ye, YeT, Zte);
    k_down  <<<dim3(BH, Nn / 16), 32, 0, stream>>>(Zte, Ye, Xe, dnB);
    k_main  <<<dim3(BH, Dd), 256, mainSmem, stream>>>(Xe, YeT, Zte, v1bf, v2bf, upB);
    k_out   <<<(Bb * Nn * Cc) / 256, 256, 0, stream>>>(upB, dnB, out);
}
